// GATLayer_21706764714525
// MI455X (gfx1250) — compile-verified, matrix operand order checked by reading, not run on hardware
//
#include <hip/hip_runtime.h>
#include <math.h>

// ---------------------------------------------------------------------------
// GAT 2-layer pipeline for MI455X (gfx1250, wave32).
// Dense GEMMs use v_wmma_f32_16x16x32_f16 (f16 in, f32 accumulate); W is
// pre-packed in LDS into the exact B-fragment layout so the hot loop issues
// ds_load_b128 only. Graph softmax/aggregation uses native fp32 global
// atomics (all per-node state ~100MB is resident in the 192MB L2).
// ---------------------------------------------------------------------------

#define NN   50000          // nodes
#define EE   600000         // edges (before self loops)
#define TOTE (EE + NN)      // edges + self loops
#define DD   128
#define EPS  1e-5f
#define NEG_SLOPE 0.2f

typedef __attribute__((ext_vector_type(16))) _Float16 v16h;
typedef __attribute__((ext_vector_type(8)))  float    v8f;

__device__ __forceinline__ void atomAddF(float* p, float v) {
  __hip_atomic_fetch_add(p, v, __ATOMIC_RELAXED, __HIP_MEMORY_SCOPE_AGENT);
}
__device__ __forceinline__ void atomMaxF(float* p, float v) {
  __hip_atomic_fetch_max(p, v, __ATOMIC_RELAXED, __HIP_MEMORY_SCOPE_AGENT);
}

// ---------------------------------------------------------------------------
// H[N,128] = X[N,128] @ W[128,128] via WMMA f16->f32.
// Block = 256 threads (8 waves) processes 128 rows; wave w owns cols [16w,16w+16).
// W is staged once per block into LDS *pre-packed in B-fragment order*:
//   sWp[((tile*4 + kt)*32 + lane)*16 + i]  (32B contiguous per lane fragment)
// so each B fragment is two ds_load_b128. The 4 B fragments live in registers
// and are reused across all 8 row-tiles (32 WMMA per wave).
// ---------------------------------------------------------------------------
__global__ __launch_bounds__(256)
void gat_gemm_wmma(const float* __restrict__ X, const float* __restrict__ W,
                   float* __restrict__ H, int nrows) {
  __shared__ _Float16 sWp[DD * DD];    // packed B fragments, 32 KB
  __shared__ _Float16 sX[128 * DD];    // [r][k] row-major,    32 KB

  const int tid   = threadIdx.x;
  const int mBase = blockIdx.x * 128;

  // Stage W (f32 -> f16), packed: idx = ((tile*4 + kt)*32 + lane)*16 + i.
  // B 32x16 f16 fragment layout (ISA 7.12.2, symmetric to A with N=lane%16):
  //   N = tile*16 + lane%16
  //   K = kt*32 + (lane>=16 ? 8 : 0) + (i<8 ? i : 16 + i-8)
  for (int idx = tid; idx < DD * DD; idx += 256) {
    int i    = idx & 15;
    int lane = (idx >> 4) & 31;
    int kt   = (idx >> 9) & 3;
    int tile = idx >> 11;
    int n = tile * 16 + (lane & 15);
    int k = kt * 32 + (lane >> 4) * 8 + (i & 7) + ((i >> 3) << 4);
    sWp[idx] = (_Float16)W[k * DD + n];
  }
  // Stage 128-row X stripe (zero-pad rows past nrows).
  for (int idx = tid; idx < 128 * DD; idx += 256) {
    int r = idx >> 7, k = idx & 127;
    int gr = mBase + r;
    sX[idx] = (gr < nrows) ? (_Float16)X[(size_t)gr * DD + k] : (_Float16)0.0f;
  }
  __syncthreads();

  const int lane = tid & 31;
  const int wave = tid >> 5;
  const int n0   = wave * 16;
  const int lr   = lane & 15;          // A: row M; B/C/D: col N
  const int hi   = (lane >> 4) & 1;    // lane group selects K sub-range

  // B fragments: one contiguous 32B LDS load each (-> 2x ds_load_b128).
  v16h bfrag[4];
#pragma unroll
  for (int kt = 0; kt < 4; ++kt)
    bfrag[kt] = *(const v16h*)&sWp[(((wave * 4 + kt) * 32) + lane) * 16];

  for (int rt = 0; rt < 8; ++rt) {     // row-tiles; guard is wave-uniform
    int m0 = mBase + rt * 16;
    if (m0 >= nrows) break;            // (nrows % 16 == 0)
    v8f acc = {};
#pragma unroll
    for (int kt = 0; kt < 4; ++kt) {
      const int kb = kt * 32 + hi * 8;
      const _Float16* ar = &sX[(rt * 16 + lr) * DD + kb];
      v16h a;
#pragma unroll
      for (int i = 0; i < 8; ++i) { a[i] = ar[i]; a[i + 8] = ar[16 + i]; }
      acc = __builtin_amdgcn_wmma_f32_16x16x32_f16(
          /*neg_a=*/false, a, /*neg_b=*/false, bfrag[kt],
          /*c_mod=*/(short)0, acc, /*reuse_a=*/false, /*reuse_b=*/false);
    }
    // C/D f32 16x16: VGPR v -> M = v + (lane<16 ? 0 : 8), N = lane%16.
    const int rbase = (lane < 16) ? 0 : 8;
#pragma unroll
    for (int v = 0; v < 8; ++v)
      H[(size_t)(m0 + rbase + v) * DD + n0 + lr] = acc[v];
  }
}

// ---------------------------------------------------------------------------
// Per-node attention logits: e_src[n,h] = <H[n, h*hd:(h+1)*hd], a_src[h]>.
// ---------------------------------------------------------------------------
__global__ void gat_attn_logits(const float* __restrict__ H,
                                const float* __restrict__ a_src,
                                const float* __restrict__ a_dst,
                                float* __restrict__ esrc,
                                float* __restrict__ edst,
                                int heads, int hd) {
  int t = blockIdx.x * blockDim.x + threadIdx.x;
  if (t >= NN * heads) return;
  int node = t / heads, hh = t - node * heads;
  const float* base = H + (size_t)node * DD + hh * hd;
  const float* as = a_src + hh * hd;
  const float* ad = a_dst + hh * hd;
  float s0 = 0.f, s1 = 0.f;
  for (int c = 0; c < hd; ++c) { float v = base[c]; s0 += v * as[c]; s1 += v * ad[c]; }
  esrc[t] = s0; edst[t] = s1;
}

__global__ void gat_init_softmax(float* __restrict__ m, float* __restrict__ denom, int n) {
  int t = blockIdx.x * blockDim.x + threadIdx.x;
  if (t < n) { m[t] = -__builtin_huge_valf(); denom[t] = 0.f; }
}

__global__ void gat_zero(float* __restrict__ p, int n) {
  int t = blockIdx.x * blockDim.x + threadIdx.x;
  if (t < n) p[t] = 0.f;
}

__device__ __forceinline__ void edge_endpoints(const int* __restrict__ ei, int e,
                                               int& src, int& dst) {
  if (e < EE) { src = ei[e]; dst = ei[EE + e]; }   // edge_index rows 0/1
  else        { src = dst = e - EE; }              // implicit self loop
}

__device__ __forceinline__ float leaky(float v) {
  return v > 0.f ? v : NEG_SLOPE * v;
}

// Pass 1: segment max of leaky_relu(e_src[src]+e_dst[dst]) per (dst, head).
__global__ void gat_edge_max(const int* __restrict__ ei,
                             const float* __restrict__ esrc,
                             const float* __restrict__ edst,
                             float* __restrict__ m, int heads) {
  int t = blockIdx.x * blockDim.x + threadIdx.x;
  if (t >= TOTE * heads) return;
  int e = t / heads, hh = t - e * heads;
  int src, dst; edge_endpoints(ei, e, src, dst);
  float v = leaky(esrc[src * heads + hh] + edst[dst * heads + hh]);
  atomMaxF(&m[dst * heads + hh], v);
}

// Pass 2: alpha = exp(e - m[dst]); denom[dst] += alpha.
__global__ void gat_edge_expsum(const int* __restrict__ ei,
                                const float* __restrict__ esrc,
                                const float* __restrict__ edst,
                                const float* __restrict__ m,
                                float* __restrict__ alpha,
                                float* __restrict__ denom, int heads) {
  int t = blockIdx.x * blockDim.x + threadIdx.x;
  if (t >= TOTE * heads) return;
  int e = t / heads, hh = t - e * heads;
  int src, dst; edge_endpoints(ei, e, src, dst);
  float v  = leaky(esrc[src * heads + hh] + edst[dst * heads + hh]);
  float al = __expf(v - m[dst * heads + hh]);
  alpha[t] = al;
  atomAddF(&denom[dst * heads + hh], al);
}

// Pass 3: agg[dst] += (alpha/denom[dst]) * H[src]. One wave32 per edge; each
// lane covers 4 of the 128 feature channels. Gather + atomic scatter both hit
// L2 (feature matrices are 25.6MB each, well inside the 192MB L2). The edge
// list is the only real HBM stream -> prefetch it ahead.
__global__ __launch_bounds__(256)
void gat_edge_aggregate(const int* __restrict__ ei,
                        const float* __restrict__ alpha,
                        const float* __restrict__ denom,
                        const float* __restrict__ H,
                        float* __restrict__ agg, int heads, int hd) {
  int wave = (blockIdx.x * blockDim.x + threadIdx.x) >> 5;
  int lane = threadIdx.x & 31;
  if (wave >= TOTE) return;
  // Prefetch edge indices ~8K edges ahead (global_prefetch_b8).
  if (lane == 0 && wave + 8192 < EE) {
    __builtin_prefetch(&ei[wave + 8192], 0, 1);
    __builtin_prefetch(&ei[EE + wave + 8192], 0, 1);
  }
  int src, dst; edge_endpoints(ei, wave, src, dst);
  const float* hs = H + (size_t)src * DD;
  float* ag       = agg + (size_t)dst * DD;
#pragma unroll
  for (int f = lane; f < DD; f += 32) {
    int hh = f / hd;
    float coef = alpha[wave * heads + hh] / denom[dst * heads + hh];
    atomAddF(&ag[f], coef * hs[f]);
  }
}

// out = agg + bias[col] + residual
__global__ void gat_bias_residual(const float* __restrict__ agg,
                                  const float* __restrict__ bias,
                                  const float* __restrict__ res,
                                  float* __restrict__ out) {
  int t = blockIdx.x * blockDim.x + threadIdx.x;
  if (t >= NN * DD) return;
  out[t] = agg[t] + bias[t & (DD - 1)] + res[t];
}

// Column-wise partial sums for BatchNorm: block = 128 threads (one per column),
// each block reduces a 256-row stripe, then one atomicAdd per (block, column).
__global__ __launch_bounds__(128)
void gat_colstats(const float* __restrict__ X, float* __restrict__ sum,
                  float* __restrict__ sumsq) {
  int col = threadIdx.x;
  int r0 = blockIdx.x * 256;
  int r1 = min(r0 + 256, NN);
  float s = 0.f, q = 0.f;
  for (int r = r0; r < r1; ++r) {
    float v = X[(size_t)r * DD + col];
    s += v; q += v * v;
  }
  atomAddF(&sum[col], s);
  atomAddF(&sumsq[col], q);
}

__global__ void gat_bn_apply(const float* __restrict__ X,
                             const float* __restrict__ sum,
                             const float* __restrict__ sumsq,
                             const float* __restrict__ gamma,
                             const float* __restrict__ beta,
                             float* __restrict__ out, int do_relu) {
  int t = blockIdx.x * blockDim.x + threadIdx.x;
  if (t >= NN * DD) return;
  int col = t & (DD - 1);
  const float invN = 1.0f / (float)NN;
  float mean = sum[col] * invN;
  float var  = sumsq[col] * invN - mean * mean;
  float y = (X[t] - mean) * rsqrtf(var + EPS) * gamma[col] + beta[col];
  if (do_relu) y = fmaxf(y, 0.f);
  out[t] = y;
}

// ---------------------------------------------------------------------------
extern "C" void kernel_launch(void* const* d_in, const int* in_sizes, int n_in,
                              void* d_out, int out_size, void* d_ws, size_t ws_size,
                              hipStream_t stream) {
  const float* x       = (const float*)d_in[0];
  const int*   ei      = (const int*)  d_in[1];   // [2, EE]
  const float* W1      = (const float*)d_in[2];
  const float* a_src1  = (const float*)d_in[3];
  const float* a_dst1  = (const float*)d_in[4];
  const float* b1      = (const float*)d_in[5];
  const float* W2      = (const float*)d_in[6];
  const float* a_src2  = (const float*)d_in[7];
  const float* a_dst2  = (const float*)d_in[8];
  const float* b2      = (const float*)d_in[9];
  const float* bng     = (const float*)d_in[10];
  const float* bnb     = (const float*)d_in[11];
  float* out = (float*)d_out;

  // Workspace carve-out (~91 MB total).
  size_t off = 0;
  auto carve = [&](size_t bytes) {
    void* p = (char*)d_ws + off;
    off += (bytes + 255) & ~(size_t)255;
    return p;
  };
  float* h      = (float*)carve((size_t)NN * DD * 4);   // GEMM output (per layer)
  float* agg    = (float*)carve((size_t)NN * DD * 4);   // aggregation accumulator
  float* y1     = (float*)carve((size_t)NN * DD * 4);   // layer-1 output / residual
  float* esrc   = (float*)carve((size_t)NN * 4 * 4);
  float* edst   = (float*)carve((size_t)NN * 4 * 4);
  float* mbuf   = (float*)carve((size_t)NN * 4 * 4);
  float* denom  = (float*)carve((size_t)NN * 4 * 4);
  float* alpha  = (float*)carve((size_t)TOTE * 4 * 4);
  float* colsum = (float*)carve(DD * 4);
  float* colsq  = (float*)carve(DD * 4);

  const int B = 256;
  const int gFeat = (NN * DD + B - 1) / B;       // element-wise over N*D
  const int gWave = (TOTE + 7) / 8;              // 1 wave32 per edge
  const int gStat = (NN + 255) / 256;
  const int gGemm = (NN + 127) / 128;            // 128 rows per block

  auto run_layer = [&](const float* Xin, const float* W, const float* as,
                       const float* ad, const float* bias, const float* res,
                       float* dest, int heads, int do_relu) {
    const int hd = DD / heads;
    gat_gemm_wmma<<<gGemm, 256, 0, stream>>>(Xin, W, h, NN);
    gat_attn_logits<<<(NN * heads + B - 1) / B, B, 0, stream>>>(h, as, ad, esrc, edst, heads, hd);
    gat_init_softmax<<<(NN * heads + B - 1) / B, B, 0, stream>>>(mbuf, denom, NN * heads);
    gat_edge_max<<<(TOTE * heads + B - 1) / B, B, 0, stream>>>(ei, esrc, edst, mbuf, heads);
    gat_edge_expsum<<<(TOTE * heads + B - 1) / B, B, 0, stream>>>(ei, esrc, edst, mbuf, alpha, denom, heads);
    gat_zero<<<gFeat, B, 0, stream>>>(agg, NN * DD);
    gat_edge_aggregate<<<gWave, B, 0, stream>>>(ei, alpha, denom, h, agg, heads, hd);
    gat_bias_residual<<<gFeat, B, 0, stream>>>(agg, bias, res, h);  // h := pre-BN
    gat_zero<<<1, DD, 0, stream>>>(colsum, DD);
    gat_zero<<<1, DD, 0, stream>>>(colsq, DD);
    gat_colstats<<<gStat, DD, 0, stream>>>(h, colsum, colsq);
    gat_bn_apply<<<gFeat, B, 0, stream>>>(h, colsum, colsq, bng, bnb, dest, do_relu);
  };

  // Layer 1: GATConv(128 -> 4 heads x 32) + residual(x) + BN + ReLU
  run_layer(x, W1, a_src1, a_dst1, b1, x, y1, /*heads=*/4, /*relu=*/1);
  // Layer 2: GATConv(128 -> 1 head x 128) + residual(y1) + BN
  run_layer(y1, W2, a_src2, a_dst2, b2, y1, out, /*heads=*/1, /*relu=*/0);
}
